// PointTransformerCls_83640193122341
// MI455X (gfx1250) — compile-verified, matrix operand order checked by reading, not run on hardware
//
#include <hip/hip_runtime.h>
#include <hip/hip_bf16.h>
#include <math.h>

// ---------------------------------------------------------------------------
// PointTransformerCls forward for MI455X (gfx1250), wave32 + WMMA.
// Dense linears run through v_wmma_f32_16x16x32_f16. Weights are pre-
// transposed/converted to padded f16 (one tiny kernel per GEMM) so the B tile
// fill is pure b128 traffic; A tiles use a block-uniform fast path with
// float4 loads. Attention inner tensors are processed in 2048-row chunks so
// [rows*K,256] scratch (~33MB) stays resident in the 192MB L2.
//
// Input pointer order (setup_inputs dict order; params flattened jax-style:
// dict keys alphabetical, lists in order, each linear {'b','w'} -> b then w):
//   d_in[0]=xyz (8,2048,3) f32   d_in[1]=knn_idx (8,2048,16) i32
//   then params: fc2a(b,w) fc2b fc2c fct1 fct2 sa1[0..2](b,w) sa2[..] sa3[..]
//   tr1..tr4 each: delta1(b,w) delta2 fc1 fc2 gamma1 gamma2 wk wq wv
// ---------------------------------------------------------------------------

typedef __attribute__((ext_vector_type(16))) _Float16 v16h;
typedef __attribute__((ext_vector_type(8)))  _Float16 v8h;
typedef __attribute__((ext_vector_type(4)))  _Float16 v4h;
typedef __attribute__((ext_vector_type(8)))  float    v8f;

constexpr int kB = 8;        // batch
constexpr int kN = 2048;     // points
constexpr int kK = 16;       // knn
constexpr int kD = 256;      // d_model
constexpr int kCH = 2048;    // chunk of (b,n) rows for attention scratch

// --------------------- weight re-layout: f32 -> padded f16 -----------------
// Wt[c*Kdp + k] = (c<Ncol && k<Kd) ? (f16)W[k*Ncol+c] : 0
__global__ void pt_wcast(const float* __restrict__ W, _Float16* __restrict__ Wt,
                         int Kd, int Ncol, int Kdp, int Ncolp)
{
    const int tid = blockIdx.x * blockDim.x + threadIdx.x;
    if (tid >= Ncolp * Kdp) return;
    const int k = tid % Kdp;
    const int c = tid / Kdp;
    float v = 0.f;
    if (k < Kd && c < Ncol) v = W[(size_t)k * Ncol + c];
    Wt[tid] = (_Float16)v;
}

// ------------------------------ WMMA GEMM ----------------------------------
// C[M,Ncol] = act(A[M,Kd] @ W[Kd,Ncol] + bias) + resid
// Block: 128 threads = 4 waves, 64x64 output tile. Wave w owns rows
// [tm*64+w*16, +16), computes 4 16x16 subtiles across N with one A fragment.
__global__ __launch_bounds__(128)
void pt_gemm_wmma(const float* __restrict__ A,
                  const _Float16* __restrict__ Wt,   // [Ncolp][Kdp] f16
                  const float* __restrict__ bias,
                  const float* __restrict__ resid,
                  float* __restrict__ C,
                  int M, int Kd, int Ncol, int Kdp, int relu)
{
    __shared__ alignas(16) _Float16 sA[64 * 32];   // [row][k]
    __shared__ alignas(16) _Float16 sB[64 * 32];   // [col][k]

    const int tid    = threadIdx.x;        // 0..127
    const int lane   = tid & 31;
    const int wave   = tid >> 5;           // 0..3
    const int tilesN = (Ncol + 63) >> 6;
    const int tm     = blockIdx.x / tilesN;
    const int tn     = blockIdx.x % tilesN;
    const int row0   = tm << 6;
    const int col0   = tn << 6;

    const int mr    = lane & 15;           // A row / B col within subtile
    const int khalf = (lane >> 4) << 3;    // 0 or 8

    // B fill coordinates (constant per thread): half a row of sB each
    const int bc = tid >> 1;               // tile col 0..63
    const int bk = (tid & 1) << 4;         // k offset 0 or 16

    v8f acc0 = {}, acc1 = {}, acc2 = {}, acc3 = {};

    for (int k0 = 0; k0 < Kd; k0 += 32) {
        // ---- A tile fill: fast path is unconditional float4 loads ----
        const bool fastA = (row0 + 64 <= M) && (k0 + 32 <= Kd) && ((Kd & 3) == 0);
        if (fastA) {
#pragma unroll
            for (int i = 0; i < 4; ++i) {
                const int e  = (tid + (i << 7)) << 2;   // element, multiple of 4
                const int r  = e >> 5;
                const int kk = e & 31;
                const float4 f =
                    *(const float4*)(A + (size_t)(row0 + r) * Kd + (k0 + kk));
                v4h h;
                h[0] = (_Float16)f.x; h[1] = (_Float16)f.y;
                h[2] = (_Float16)f.z; h[3] = (_Float16)f.w;
                *(v4h*)(sA + e) = h;
            }
        } else {
#pragma unroll
            for (int i = 0; i < 16; ++i) {
                const int idx = tid + (i << 7);
                const int r   = idx >> 5;
                const int kk  = idx & 31;
                const int gr  = row0 + r;
                const int gk  = k0 + kk;
                const int cr  = gr < M ? gr : (M - 1);
                const int ck  = gk < Kd ? gk : (Kd - 1);
                const float m = (gr < M && gk < Kd) ? 1.f : 0.f;
                sA[idx] = (_Float16)(A[(size_t)cr * Kd + ck] * m);  // mask mul,
            }                                   // load always used -> no exec
        }
        // ---- B tile fill: pure b128, padded f16 source, no checks ----
        {
            const _Float16* wsrc = Wt + (size_t)(col0 + bc) * Kdp + k0 + bk;
            *(v8h*)(sB + bc * 32 + bk)     = *(const v8h*)(wsrc);
            *(v8h*)(sB + bc * 32 + bk + 8) = *(const v8h*)(wsrc + 8);
        }
        __syncthreads();

        // ---- fragments from LDS: two b128 reads each, per ISA layout ----
        const _Float16* arow = sA + ((wave << 4) + mr) * 32;
        const v8h alo = *(const v8h*)(arow + khalf);
        const v8h ahi = *(const v8h*)(arow + 16 + khalf);
        const v16h af = __builtin_shufflevector(alo, ahi,
            0, 1, 2, 3, 4, 5, 6, 7, 8, 9, 10, 11, 12, 13, 14, 15);

        const _Float16* b0 = sB + (0 * 16 + mr) * 32;
        const _Float16* b1 = sB + (1 * 16 + mr) * 32;
        const _Float16* b2 = sB + (2 * 16 + mr) * 32;
        const _Float16* b3 = sB + (3 * 16 + mr) * 32;
        const v16h bf0 = __builtin_shufflevector(
            *(const v8h*)(b0 + khalf), *(const v8h*)(b0 + 16 + khalf),
            0, 1, 2, 3, 4, 5, 6, 7, 8, 9, 10, 11, 12, 13, 14, 15);
        const v16h bf1 = __builtin_shufflevector(
            *(const v8h*)(b1 + khalf), *(const v8h*)(b1 + 16 + khalf),
            0, 1, 2, 3, 4, 5, 6, 7, 8, 9, 10, 11, 12, 13, 14, 15);
        const v16h bf2 = __builtin_shufflevector(
            *(const v8h*)(b2 + khalf), *(const v8h*)(b2 + 16 + khalf),
            0, 1, 2, 3, 4, 5, 6, 7, 8, 9, 10, 11, 12, 13, 14, 15);
        const v16h bf3 = __builtin_shufflevector(
            *(const v8h*)(b3 + khalf), *(const v8h*)(b3 + 16 + khalf),
            0, 1, 2, 3, 4, 5, 6, 7, 8, 9, 10, 11, 12, 13, 14, 15);

        acc0 = __builtin_amdgcn_wmma_f32_16x16x32_f16(false, af, false, bf0,
                                                      (short)0, acc0, false, false);
        acc1 = __builtin_amdgcn_wmma_f32_16x16x32_f16(false, af, false, bf1,
                                                      (short)0, acc1, false, false);
        acc2 = __builtin_amdgcn_wmma_f32_16x16x32_f16(false, af, false, bf2,
                                                      (short)0, acc2, false, false);
        acc3 = __builtin_amdgcn_wmma_f32_16x16x32_f16(false, af, false, bf3,
                                                      (short)0, acc3, false, false);
        __syncthreads();
    }

    // ---- epilogue: bias / relu / residual, bounds-checked stores ----
    const int rbase = row0 + (wave << 4) + ((lane >> 4) ? 8 : 0);
    v8f accs[4] = { acc0, acc1, acc2, acc3 };
#pragma unroll
    for (int nb = 0; nb < 4; ++nb) {
        const int col = col0 + (nb << 4) + mr;
        if (col >= Ncol) continue;
        const float bv = bias ? bias[col] : 0.f;
#pragma unroll
        for (int r = 0; r < 8; ++r) {
            const int row = rbase + r;
            if (row < M) {
                float v = accs[nb][r] + bv;
                if (relu)  v = v > 0.f ? v : 0.f;
                const size_t o = (size_t)row * Ncol + col;
                if (resid) v += resid[o];
                C[o] = v;
            }
        }
    }
}

static void launch_gemm(hipStream_t s, const float* A, const float* W,
                        const float* bias, const float* resid, float* C,
                        int M, int Kd, int Ncol, int relu, _Float16* wt)
{
    const int Kdp   = (Kd + 31) & ~31;
    const int Ncolp = (Ncol + 63) & ~63;
    const int wtot  = Ncolp * Kdp;
    pt_wcast<<<(wtot + 255) / 256, 256, 0, s>>>(W, wt, Kd, Ncol, Kdp, Ncolp);
    const int tiles = ((M + 63) / 64) * (Ncolp >> 6);
    pt_gemm_wmma<<<tiles, 128, 0, s>>>(A, wt, bias, resid, C, M, Kd, Ncol,
                                       Kdp, relu);
}

// --------------------------- elementwise kernels ---------------------------

__global__ void pt_xyz_diff(const float* __restrict__ xyz,
                            const int* __restrict__ idx,
                            float* __restrict__ out,
                            int bn0, int rows, int Nn, int Kk)
{
    const int tid = blockIdx.x * blockDim.x + threadIdx.x;
    const int tot = rows * Kk * 3;
    if (tid >= tot) return;
    const int c  = tid % 3;
    const int rk = tid / 3;
    const int k  = rk % Kk;
    const int r  = rk / Kk;
    const int bn = bn0 + r;
    const int b  = bn / Nn;
    const int id = idx[(size_t)bn * Kk + k];
    out[tid] = xyz[(size_t)bn * 3 + c] - xyz[((size_t)b * Nn + id) * 3 + c];
}

__global__ void pt_attn_input(const float* __restrict__ q,
                              const float* __restrict__ kfull,
                              const float* __restrict__ pos,
                              const int* __restrict__ idx,
                              float* __restrict__ out,
                              int bn0, int rows, int Nn, int Kk, int C)
{
    const int tid = blockIdx.x * blockDim.x + threadIdx.x;
    const int tot = rows * Kk * C;
    if (tid >= tot) return;
    const int c  = tid % C;
    const int rk = tid / C;
    const int k  = rk % Kk;
    const int r  = rk / Kk;
    const int bn = bn0 + r;
    const int b  = bn / Nn;
    const int id = idx[(size_t)bn * Kk + k];
    out[tid] = q[(size_t)bn * C + c]
             - kfull[((size_t)b * Nn + id) * C + c]
             + pos[tid];
}

__global__ void pt_softmax_k(float* __restrict__ attn,
                             int rows, int Kk, int C, float scale)
{
    const int tid = blockIdx.x * blockDim.x + threadIdx.x;
    if (tid >= rows * C) return;
    const int c = tid % C;
    const int r = tid / C;
    float* base = attn + (size_t)r * Kk * C + c;
    float mx = -3.4e38f;
    for (int k = 0; k < Kk; ++k) {
        const float v = base[(size_t)k * C] * scale;
        if (v > mx) mx = v;
    }
    float sum = 0.f;
    for (int k = 0; k < Kk; ++k) {
        const float e = expf(base[(size_t)k * C] * scale - mx);
        base[(size_t)k * C] = e;
        sum += e;
    }
    const float inv = 1.f / sum;
    for (int k = 0; k < Kk; ++k) base[(size_t)k * C] *= inv;
}

__global__ void pt_attn_reduce(const float* __restrict__ attn,
                               const float* __restrict__ vfull,
                               const float* __restrict__ pos,
                               const int* __restrict__ idx,
                               float* __restrict__ res,
                               int bn0, int rows, int Nn, int Kk, int C)
{
    const int tid = blockIdx.x * blockDim.x + threadIdx.x;
    if (tid >= rows * C) return;
    const int c  = tid % C;
    const int r  = tid / C;
    const int bn = bn0 + r;
    const int b  = bn / Nn;
    float acc = 0.f;
    for (int k = 0; k < Kk; ++k) {
        const int id   = idx[(size_t)bn * Kk + k];
        const size_t o = ((size_t)r * Kk + k) * C + c;
        acc += attn[o] * (vfull[((size_t)b * Nn + id) * C + c] + pos[o]);
    }
    res[(size_t)bn * C + c] = acc;
}

// --------------------------- sampling / grouping ---------------------------

__global__ void pt_fps(const float* __restrict__ xyz, int* __restrict__ fidx,
                       int Nn, int npoint)
{
    __shared__ float dist[kN];
    __shared__ float rv[256];
    __shared__ int   ri[256];
    __shared__ int   s_far;
    const int b  = blockIdx.x;
    const int t  = threadIdx.x;
    const int nt = blockDim.x;
    const float* xb = xyz + (size_t)b * Nn * 3;
    for (int i = t; i < Nn; i += nt) dist[i] = 1e10f;
    if (t == 0) s_far = 0;
    __syncthreads();
    for (int it = 0; it < npoint; ++it) {
        const int far = s_far;
        if (t == 0) fidx[(size_t)b * npoint + it] = far;
        const float cx = xb[far * 3 + 0];
        const float cy = xb[far * 3 + 1];
        const float cz = xb[far * 3 + 2];
        float bestv = -1.f; int besti = 0;
        for (int i = t; i < Nn; i += nt) {
            const float dx = xb[i * 3 + 0] - cx;
            const float dy = xb[i * 3 + 1] - cy;
            const float dz = xb[i * 3 + 2] - cz;
            const float d  = dx * dx + dy * dy + dz * dz;
            const float nd = fminf(dist[i], d);
            dist[i] = nd;
            if (nd > bestv) { bestv = nd; besti = i; }
        }
        rv[t] = bestv; ri[t] = besti;
        __syncthreads();
        for (int s = nt >> 1; s > 0; s >>= 1) {
            if (t < s) {
                if (rv[t + s] > rv[t] ||
                    (rv[t + s] == rv[t] && ri[t + s] < ri[t])) {
                    rv[t] = rv[t + s]; ri[t] = ri[t + s];
                }
            }
            __syncthreads();
        }
        if (t == 0) s_far = ri[0];
        __syncthreads();
    }
}

__global__ void pt_gather_xyz(const float* __restrict__ xyz,
                              const int* __restrict__ idx,
                              float* __restrict__ out, int Nn, int np)
{
    const int tid = blockIdx.x * blockDim.x + threadIdx.x;
    if (tid >= kB * np * 3) return;
    const int c  = tid % 3;
    const int bj = tid / 3;
    const int b  = bj / np;
    const int id = idx[bj];
    out[tid] = xyz[((size_t)b * Nn + id) * 3 + c];
}

__global__ void pt_ball(const float* __restrict__ xyz,
                        const float* __restrict__ ctr,
                        int* __restrict__ out,
                        int Nn, int np, int ns, float r2)
{
    const int tid = blockIdx.x * blockDim.x + threadIdx.x;
    if (tid >= kB * np) return;
    const int b = tid / np;
    const float* xb = xyz + (size_t)b * Nn * 3;
    const float cx = ctr[tid * 3 + 0];
    const float cy = ctr[tid * 3 + 1];
    const float cz = ctr[tid * 3 + 2];
    int* o = out + (size_t)tid * ns;
    int cnt = 0, first = -1;
    for (int n = 0; n < Nn && cnt < ns; ++n) {
        const float dx = xb[n * 3 + 0] - cx;
        const float dy = xb[n * 3 + 1] - cy;
        const float dz = xb[n * 3 + 2] - cz;
        if (dx * dx + dy * dy + dz * dz <= r2) {
            if (first < 0) first = n;
            o[cnt++] = n;
        }
    }
    if (first < 0) first = 0;
    for (; cnt < ns; ++cnt) o[cnt] = first;
}

__global__ void pt_group(const float* __restrict__ xyz,
                         const float* __restrict__ feats,
                         const float* __restrict__ ctr,
                         const int* __restrict__ idx,
                         float* __restrict__ out,
                         int Nn, int np, int ns, int C)
{
    const int tid = blockIdx.x * blockDim.x + threadIdx.x;
    if (tid >= kB * np * ns) return;
    const int s  = tid % ns;
    const int bj = tid / ns;
    const int b  = bj / np;
    const int id = idx[(size_t)bj * ns + s];
    const float* src = xyz + ((size_t)b * Nn + id) * 3;
    const float* cc  = ctr + (size_t)bj * 3;
    float* o = out + (size_t)tid * (3 + C);
    o[0] = src[0] - cc[0];
    o[1] = src[1] - cc[1];
    o[2] = src[2] - cc[2];
    const float* f = feats + ((size_t)b * Nn + id) * C;
    for (int c = 0; c < C; ++c) o[3 + c] = f[c];
}

__global__ void pt_maxpool(const float* __restrict__ in,
                           float* __restrict__ out, int BJ, int ns, int C)
{
    const int tid = blockIdx.x * blockDim.x + threadIdx.x;
    if (tid >= BJ * C) return;
    const int c  = tid % C;
    const int bj = tid / C;
    float m = -3.4e38f;
    for (int s = 0; s < ns; ++s)
        m = fmaxf(m, in[((size_t)bj * ns + s) * C + c]);
    out[tid] = m;
}

__global__ void pt_knn(const float* __restrict__ xyz, int* __restrict__ out,
                       int Nn)
{
    const int tid = blockIdx.x * blockDim.x + threadIdx.x;
    if (tid >= kB * Nn) return;
    const int b = tid / Nn;
    const int n = tid % Nn;
    const float* xb = xyz + (size_t)b * Nn * 3;
    const float px = xb[n * 3 + 0], py = xb[n * 3 + 1], pz = xb[n * 3 + 2];
    float bd[16]; int bi[16];
#pragma unroll
    for (int i = 0; i < 16; ++i) { bd[i] = 3.4e38f; bi[i] = 0; }
    for (int m = 0; m < Nn; ++m) {
        const float dx = xb[m * 3 + 0] - px;
        const float dy = xb[m * 3 + 1] - py;
        const float dz = xb[m * 3 + 2] - pz;
        const float d  = dx * dx + dy * dy + dz * dz;
        if (d < bd[15]) {
            int p = 15;
            while (p > 0 && bd[p - 1] > d) {
                bd[p] = bd[p - 1]; bi[p] = bi[p - 1]; --p;
            }
            bd[p] = d; bi[p] = m;
        }
    }
    for (int i = 0; i < 16; ++i) out[(size_t)tid * 16 + i] = bi[i];
}

__global__ void pt_concat(const float* __restrict__ xyz,
                          const float* __restrict__ feats,
                          float* __restrict__ out, int Ns, int C)
{
    const int tid = blockIdx.x * blockDim.x + threadIdx.x;
    const int tot = kB * Ns * (3 + C);
    if (tid >= tot) return;
    const int c  = tid % (3 + C);
    const int bs = tid / (3 + C);
    out[tid] = (c < 3) ? xyz[(size_t)bs * 3 + c]
                       : feats[(size_t)bs * C + (c - 3)];
}

__global__ void pt_fill_f(float* p, float v, int n)
{ const int i = blockIdx.x * blockDim.x + threadIdx.x; if (i < n) p[i] = v; }

__global__ void pt_fill_i(int* p, int v, int n)
{ const int i = blockIdx.x * blockDim.x + threadIdx.x; if (i < n) p[i] = v; }

__global__ void pt_sigmoid(const float* __restrict__ in,
                           float* __restrict__ out, int n)
{
    const int i = blockIdx.x * blockDim.x + threadIdx.x;
    if (i < n) out[i] = 0.4f + 0.6f / (1.f + expf(-in[i]));
}

// --------------------------- transformer driver ----------------------------

struct TrP {
    const float *d1b, *d1w, *d2b, *d2w, *fc1b, *fc1w, *fc2b, *fc2w;
    const float *g1b, *g1w, *g2b, *g2w, *wk, *wq, *wv;
};

static void run_transformer(hipStream_t s, const float* xyz,
                            const float* feats, const int* knn,
                            int Nn, int keff, int dpts, const TrP& P,
                            float* mx, float* mq, float* mk, float* mv,
                            float* posin, float* bh, float* bpos, float* battn,
                            float* out, _Float16* wt)
{
    const int BN = kB * Nn;
    const int C  = kD;
    launch_gemm(s, feats, P.fc1w, P.fc1b, nullptr, mx, BN, dpts, C, 0, wt);
    launch_gemm(s, mx, P.wq, nullptr, nullptr, mq, BN, C, C, 0, wt);
    launch_gemm(s, mx, P.wk, nullptr, nullptr, mk, BN, C, C, 0, wt);
    launch_gemm(s, mx, P.wv, nullptr, nullptr, mv, BN, C, C, 0, wt);

    for (int bn0 = 0; bn0 < BN; bn0 += kCH) {
        int rows = BN - bn0; if (rows > kCH) rows = kCH;
        const int totd = rows * keff * 3;
        pt_xyz_diff<<<(totd + 255) / 256, 256, 0, s>>>(
            xyz, knn, posin, bn0, rows, Nn, keff);
        launch_gemm(s, posin, P.d1w, P.d1b, nullptr, bh,   rows * keff, 3, C, 1, wt);
        launch_gemm(s, bh,    P.d2w, P.d2b, nullptr, bpos, rows * keff, C, C, 0, wt);
        const int tota = rows * keff * C;
        pt_attn_input<<<(tota + 255) / 256, 256, 0, s>>>(
            mq, mk, bpos, knn, battn, bn0, rows, Nn, keff, C);
        launch_gemm(s, battn, P.g1w, P.g1b, nullptr, bh,    rows * keff, C, C, 1, wt);
        launch_gemm(s, bh,    P.g2w, P.g2b, nullptr, battn, rows * keff, C, C, 0, wt);
        const int tots = rows * C;
        pt_softmax_k<<<(tots + 255) / 256, 256, 0, s>>>(
            battn, rows, keff, C, 0.0625f);               // 1/sqrt(256)
        pt_attn_reduce<<<(tots + 255) / 256, 256, 0, s>>>(
            battn, mv, bpos, knn, mx, bn0, rows, Nn, keff, C); // res -> mx
    }
    launch_gemm(s, mx, P.fc2w, P.fc2b, feats, out, BN, C, dpts, 0, wt); // +pre
}

// ------------------------------- launcher ----------------------------------

extern "C" void kernel_launch(void* const* d_in, const int* in_sizes, int n_in,
                              void* d_out, int out_size, void* d_ws,
                              size_t ws_size, hipStream_t stream)
{
    (void)in_sizes; (void)n_in; (void)out_size; (void)ws_size;

    const float* xyz  = (const float*)d_in[0];
    const int*   knn1 = (const int*)d_in[1];
    int p = 2;
    auto nf = [&]() { return (const float*)d_in[p++]; };

    const float *fc2a_b = nf(), *fc2a_w = nf();
    const float *fc2b_b = nf(), *fc2b_w = nf();
    const float *fc2c_b = nf(), *fc2c_w = nf();
    const float *fct1_b = nf(), *fct1_w = nf();
    const float *fct2_b = nf(), *fct2_w = nf();
    const float *sa1b0 = nf(), *sa1w0 = nf(), *sa1b1 = nf(), *sa1w1 = nf(),
                *sa1b2 = nf(), *sa1w2 = nf();
    const float *sa2b0 = nf(), *sa2w0 = nf(), *sa2b1 = nf(), *sa2w1 = nf(),
                *sa2b2 = nf(), *sa2w2 = nf();
    const float *sa3b0 = nf(), *sa3w0 = nf(), *sa3b1 = nf(), *sa3w1 = nf(),
                *sa3b2 = nf(), *sa3w2 = nf();
    auto loadTr = [&]() {
        TrP t;
        t.d1b = nf(); t.d1w = nf(); t.d2b = nf(); t.d2w = nf();
        t.fc1b = nf(); t.fc1w = nf(); t.fc2b = nf(); t.fc2w = nf();
        t.g1b = nf(); t.g1w = nf(); t.g2b = nf(); t.g2w = nf();
        t.wk = nf(); t.wq = nf(); t.wv = nf();
        return t;
    };
    TrP trp1 = loadTr(), trp2 = loadTr(), trp3 = loadTr(), trp4 = loadTr();

    // ---- workspace bump allocator ----
    char* wsp = (char*)d_ws;
    size_t off = 0;
    auto allocF = [&](size_t n) {
        float* r = (float*)(wsp + off);
        off += ((n * sizeof(float) + 255) / 256) * 256; return r;
    };
    auto allocI = [&](size_t n) {
        int* r = (int*)(wsp + off);
        off += ((n * sizeof(int) + 255) / 256) * 256; return r;
    };
    auto allocH = [&](size_t n) {
        _Float16* r = (_Float16*)(wsp + off);
        off += ((n * sizeof(_Float16) + 255) / 256) * 256; return r;
    };

    float* mx    = allocF((size_t)kB * kN * kD);
    float* mq    = allocF((size_t)kB * kN * kD);
    float* mk    = allocF((size_t)kB * kN * kD);
    float* mv    = allocF((size_t)kB * kN * kD);
    float* posin = allocF((size_t)kCH * kK * 3);
    float* bh    = allocF((size_t)kCH * kK * kD);
    float* bpos  = allocF((size_t)kCH * kK * kD);
    float* battn = allocF((size_t)kCH * kK * kD);
    float* hid32 = allocF((size_t)kB * kN * 32);
    float* featf = allocF((size_t)kB * kN * 32);
    float* t1    = allocF((size_t)kB * kN * 32);
    float* l1x   = allocF((size_t)kB * 128 * 3);
    float* l1p   = allocF((size_t)kB * 128 * 128);
    float* t2    = allocF((size_t)kB * 128 * 128);
    float* l2x   = allocF((size_t)kB * 64 * 3);
    float* l2p   = allocF((size_t)kB * 64 * 256);
    float* t3    = allocF((size_t)kB * 64 * 256);
    float* l3x   = allocF((size_t)kB * 3);
    float* l3p   = allocF((size_t)kB * 1024);
    float* t4    = allocF((size_t)kB * 1024);
    float* h1    = allocF((size_t)kB * 256);
    float* h2    = allocF((size_t)kB * 64);
    float* h3    = allocF((size_t)kB);
    float* gpA   = allocF((size_t)kB * 64 * 64 * 131);
    float* gpB   = allocF((size_t)kB * 64 * 64 * 256);
    _Float16* wt = allocH(524288);      // max Ncolp*Kdp over all layers (sa3w2)
    int* fps1  = allocI((size_t)kB * 128);
    int* bidx1 = allocI((size_t)kB * 128 * 32);
    int* knn2  = allocI((size_t)kB * 128 * 16);
    int* fps2  = allocI((size_t)kB * 64);
    int* bidx2 = allocI((size_t)kB * 64 * 64);
    int* knn3  = allocI((size_t)kB * 64 * 16);
    int* knn4  = allocI((size_t)kB);

    // ---- stem: f = mlp2(xyz, fct1, fct2) ----
    launch_gemm(stream, xyz,   fct1_w, fct1_b, nullptr, hid32, kB * kN, 3, 32, 1, wt);
    launch_gemm(stream, hid32, fct2_w, fct2_b, nullptr, featf, kB * kN, 32, 32, 0, wt);

    // ---- tr1 (uses provided knn_idx) ----
    run_transformer(stream, xyz, featf, knn1, kN, kK, 32, trp1,
                    mx, mq, mk, mv, posin, bh, bpos, battn, t1, wt);

    // ---- sa1: fps 128, ball r=0.2 ns=32, mlps 35->64->64->128, maxpool ----
    pt_fps<<<kB, 256, 0, stream>>>(xyz, fps1, kN, 128);
    pt_gather_xyz<<<(kB * 128 * 3 + 255) / 256, 256, 0, stream>>>(
        xyz, fps1, l1x, kN, 128);
    pt_ball<<<(kB * 128 + 255) / 256, 256, 0, stream>>>(
        xyz, l1x, bidx1, kN, 128, 32, 0.04f);
    pt_group<<<(kB * 128 * 32 + 255) / 256, 256, 0, stream>>>(
        xyz, t1, l1x, bidx1, gpA, kN, 128, 32, 32);
    launch_gemm(stream, gpA, sa1w0, sa1b0, nullptr, gpB, kB * 128 * 32, 35, 64, 1, wt);
    launch_gemm(stream, gpB, sa1w1, sa1b1, nullptr, gpA, kB * 128 * 32, 64, 64, 1, wt);
    launch_gemm(stream, gpA, sa1w2, sa1b2, nullptr, gpB, kB * 128 * 32, 64, 128, 1, wt);
    pt_maxpool<<<(kB * 128 * 128 + 255) / 256, 256, 0, stream>>>(
        gpB, l1p, kB * 128, 32, 128);

    // ---- tr2 ----
    pt_knn<<<(kB * 128 + 255) / 256, 256, 0, stream>>>(l1x, knn2, 128);
    run_transformer(stream, l1x, l1p, knn2, 128, 16, 128, trp2,
                    mx, mq, mk, mv, posin, bh, bpos, battn, t2, wt);

    // ---- sa2: fps 64, ball r=0.4 ns=64, mlps 131->128->128->256 ----
    pt_fps<<<kB, 256, 0, stream>>>(l1x, fps2, 128, 64);
    pt_gather_xyz<<<(kB * 64 * 3 + 255) / 256, 256, 0, stream>>>(
        l1x, fps2, l2x, 128, 64);
    pt_ball<<<(kB * 64 + 255) / 256, 256, 0, stream>>>(
        l1x, l2x, bidx2, 128, 64, 64, 0.16f);
    pt_group<<<(kB * 64 * 64 + 255) / 256, 256, 0, stream>>>(
        l1x, t2, l2x, bidx2, gpA, 128, 64, 64, 128);
    launch_gemm(stream, gpA, sa2w0, sa2b0, nullptr, gpB, kB * 64 * 64, 131, 128, 1, wt);
    launch_gemm(stream, gpB, sa2w1, sa2b1, nullptr, gpA, kB * 64 * 64, 128, 128, 1, wt);
    launch_gemm(stream, gpA, sa2w2, sa2b2, nullptr, gpB, kB * 64 * 64, 128, 256, 1, wt);
    pt_maxpool<<<(kB * 64 * 256 + 255) / 256, 256, 0, stream>>>(
        gpB, l2p, kB * 64, 64, 256);

    // ---- tr3 ----
    pt_knn<<<(kB * 64 + 255) / 256, 256, 0, stream>>>(l2x, knn3, 64);
    run_transformer(stream, l2x, l2p, knn3, 64, 16, 256, trp3,
                    mx, mq, mk, mv, posin, bh, bpos, battn, t3, wt);

    // ---- sa3 (group_all): concat, mlps 259->256->512->1024, maxpool ----
    pt_concat<<<(kB * 64 * 259 + 255) / 256, 256, 0, stream>>>(l2x, t3, gpA, 64, 256);
    launch_gemm(stream, gpA, sa3w0, sa3b0, nullptr, gpB, kB * 64, 259, 256, 1, wt);
    launch_gemm(stream, gpB, sa3w1, sa3b1, nullptr, gpA, kB * 64, 256, 512, 1, wt);
    launch_gemm(stream, gpA, sa3w2, sa3b2, nullptr, gpB, kB * 64, 512, 1024, 1, wt);
    pt_maxpool<<<(kB * 1024 + 255) / 256, 256, 0, stream>>>(gpB, l3p, kB, 64, 1024);

    // ---- tr4: single point per batch, self-knn (K=1), zero xyz ----
    pt_fill_f<<<1, 32, 0, stream>>>(l3x, 0.f, kB * 3);
    pt_fill_i<<<1, 32, 0, stream>>>(knn4, 0, kB);
    run_transformer(stream, l3x, l3p, knn4, 1, 1, 1024, trp4,
                    mx, mq, mk, mv, posin, bh, bpos, battn, t4, wt);

    // ---- head ----
    launch_gemm(stream, t4, fc2a_w, fc2a_b, nullptr, h1, kB, 1024, 256, 1, wt);
    launch_gemm(stream, h1, fc2b_w, fc2b_b, nullptr, h2, kB, 256, 64, 1, wt);
    launch_gemm(stream, h2, fc2c_w, fc2c_b, nullptr, h3, kB, 64, 1, 0, wt);
    pt_sigmoid<<<1, 32, 0, stream>>>(h3, (float*)d_out, kB);
}
